// GNNEncoder_57836029608449
// MI455X (gfx1250) — matrix operand order repaired and checked by reading, not compiled, not run on hardware
//
#include <hip/hip_runtime.h>
#include <math.h>

// ---------------- problem constants ----------------
constexpr int BB    = 128;            // molecules
constexpr int NAT   = 64;             // atoms
constexpr int MF    = 5;              // frequencies
constexpr int EMB   = 128;
constexpr int DFF   = 2048;
constexpr int NHEAD = 4;
constexpr int HD    = 32;
constexpr int NBOND = 4;
constexpr int TT    = BB * NAT * MF;  // 40960 tokens
constexpr int BN    = BB * NAT;       // 8192 atoms total
constexpr float LN_EPS = 1e-5f;

typedef _Float16 half_t;
typedef __attribute__((ext_vector_type(16))) _Float16 v16h;
typedef __attribute__((ext_vector_type(8)))  float    v8f;
typedef unsigned int v4u __attribute__((ext_vector_type(4)));
typedef int          v8i __attribute__((ext_vector_type(8)));
typedef int          v4i __attribute__((ext_vector_type(4)));

// ---------------- WMMA helpers (gfx1250, wave32) ----------------
static __device__ __forceinline__ v8f wmma_f16(v16h a, v16h b, v8f c) {
  // v_wmma_f32_16x16x32_f16: D = A(16x32) * B(32x16) + C(16x16 f32)
  return __builtin_amdgcn_wmma_f32_16x16x32_f16(false, a, false, b, (short)0, c,
                                                false, false);
}

// A fragment (f16 LDS, [m][k] layout). Contiguous k order used here: lanes0-15 hold
// K 0..15, lanes16-31 hold K 16..31, m = lane%16.
static __device__ __forceinline__ void frag_a_mk(const half_t* S, int row0,
                                                 int k0, int stride, int lane,
                                                 v16h& a) {
  int m = row0 + (lane & 15);
  int hi = lane >> 4;
#pragma unroll
  for (int i = 0; i < 16; ++i) {
    int k = k0 + (hi << 4) + i;
    a[i] = S[m * stride + k];
  }
}
// Same pattern but from f32 LDS (TDM-staged tiles), cvt at load.
static __device__ __forceinline__ void frag_a_mk_f32(const float* S, int row0,
                                                     int k0, int stride,
                                                     int lane, v16h& a) {
  int m = row0 + (lane & 15);
  int hi = lane >> 4;
#pragma unroll
  for (int i = 0; i < 16; ++i) {
    int k = k0 + (hi << 4) + i;
    a[i] = (half_t)S[m * stride + k];
  }
}

// B fragment ([n][k] weight rows). Same contiguous k order as A: lanes0-15 K=0..15,
// lanes16-31 K=16..31, n = lane%16.
static __device__ __forceinline__ void frag_b_nk(const half_t* S, int col0,
                                                 int k0, int stride, int lane,
                                                 v16h& b) {
  int n = col0 + (lane & 15);
  int hi = lane >> 4;
#pragma unroll
  for (int i = 0; i < 16; ++i) b[i] = S[n * stride + (k0 + (hi << 4) + i)];
}
static __device__ __forceinline__ void frag_b_nk_f32(const float* S, int col0,
                                                     int k0, int stride,
                                                     int lane, v16h& b) {
  int n = col0 + (lane & 15);
  int hi = lane >> 4;
#pragma unroll
  for (int i = 0; i < 16; ++i)
    b[i] = (half_t)S[n * stride + (k0 + (hi << 4) + i)];
}

// B fragment from LDS stored [k][n] (activation layout).
static __device__ __forceinline__ void frag_b_kn(const half_t* S, int col0,
                                                 int k0, int stride, int lane,
                                                 v16h& b) {
  int n = col0 + (lane & 15);
  int hi = lane >> 4;
#pragma unroll
  for (int i = 0; i < 16; ++i) b[i] = S[(k0 + (hi << 4) + i) * stride + n];
}

// ---------------- Tensor Data Mover: 2D tile -> LDS ----------------
// D# per CDNA5 ISA. data_size=4B. Pad 1 DWORD every
// 32 DWORDs -> LDS row stride 33 dwords (bank-conflict-free gathers).
// clang-23 toolchain: 6-arg builtin (g0, g1, g2, g3, extra v8i, cpol).
static __device__ __forceinline__ void tdm_load_2d(const void* gptr,
                                                   unsigned lds_off,
                                                   int tile_h,
                                                   int row_stride_elems) {
  unsigned long long ga = (unsigned long long)gptr;
  v4u g0;
  g0[0] = 1u;                                   // count=1, user mode
  g0[1] = lds_off;                              // LDS byte address
  g0[2] = (unsigned)(ga & 0xffffffffu);         // global_addr[31:0]
  g0[3] = (unsigned)(ga >> 32) | (2u << 30);    // global_addr[56:32] | type=2
  unsigned td0 = 32u;                           // tensor_dim0 == tile width
  unsigned td1 = (unsigned)tile_h;              // tensor_dim1 == tile height
  unsigned long long s0 = (unsigned long long)row_stride_elems;
  v8i g1;
  g1[0] = (int)((2u << 16)        // data_size = 4B
                | (1u << 20)      // pad_enable
                | (4u << 22)      // pad_interval: every 32 DWORDs
                | (0u << 25));    // pad_amount: 1 DWORD
  g1[1] = (int)((td0 & 0xffffu) << 16);                  // tensor_dim0 lo16
  g1[2] = (int)((td0 >> 16) | ((td1 & 0xffffu) << 16));  // dim0 hi / dim1 lo
  g1[3] = (int)((td1 >> 16) | (32u << 16));              // dim1 hi | tile_dim0
  g1[4] = (int)((unsigned)tile_h & 0xffffu);             // tile_dim1 (dim2=0)
  g1[5] = (int)(s0 & 0xffffffffu);                       // dim0 stride lo
  g1[6] = (int)((s0 >> 32) & 0xffffu);                   // stride hi (s1=0)
  g1[7] = 0;
  v4i z4 = {0, 0, 0, 0};
  v8i z8 = {0, 0, 0, 0, 0, 0, 0, 0};
  __builtin_amdgcn_tensor_load_to_lds(g0, g1, z4, z4, z8, 0);
}

// ------- generic GEMM: out = A[M,K] @ W[N,K]^T + bias (+res), K%32==0 -------
// Block tile 64x128 (8 waves: 4 row-tiles x 2 col-halves, 16x64 per wave).
// TDM double-buffered staging, f32 tiles in LDS with 33-dword row stride.
__global__ __launch_bounds__(256) void k_gemm(const float* __restrict__ A,
                                              const float* __restrict__ W,
                                              const float* __restrict__ bias,
                                              const float* __restrict__ res,
                                              float* __restrict__ out,
                                              int Mrows, int N, int K) {
  __shared__ float Af[2][64 * 33];
  __shared__ float Bf[2][128 * 33];
  int tid = threadIdx.x, lane = tid & 31, w = tid >> 5;
  int wm = w & 3, wn = w >> 2;
  int bm = blockIdx.x * 64, bn = blockIdx.y * 128;
  int nk = K >> 5;
  if (w == 0) {
    tdm_load_2d(A + (size_t)bm * K, (unsigned)(unsigned long long)&Af[0][0],
                64, K);
    tdm_load_2d(W + (size_t)bn * K, (unsigned)(unsigned long long)&Bf[0][0],
                128, K);
  }
  v8f acc[4] = {{}, {}, {}, {}};
  for (int kc = 0; kc < nk; ++kc) {
    if (w == 0) __builtin_amdgcn_s_wait_tensorcnt(0);
    __syncthreads();  // tile kc visible; buf[kc+1 & 1] free of readers
    if (w == 0 && kc + 1 < nk) {
      int buf = (kc + 1) & 1;
      tdm_load_2d(A + (size_t)bm * K + (kc + 1) * 32,
                  (unsigned)(unsigned long long)&Af[buf][0], 64, K);
      tdm_load_2d(W + (size_t)bn * K + (kc + 1) * 32,
                  (unsigned)(unsigned long long)&Bf[buf][0], 128, K);
    }
    const float* Ab = Af[kc & 1];
    const float* Bb = Bf[kc & 1];
    v16h a, b;
    frag_a_mk_f32(Ab, wm * 16, 0, 33, lane, a);
#pragma unroll
    for (int j = 0; j < 4; ++j) {
      frag_b_nk_f32(Bb, wn * 64 + j * 16, 0, 33, lane, b);
      acc[j] = wmma_f16(a, b, acc[j]);
    }
  }
  int lane15 = lane & 15, hi = lane >> 4;
#pragma unroll
  for (int j = 0; j < 4; ++j)
#pragma unroll
    for (int r = 0; r < 8; ++r) {
      int row = bm + wm * 16 + hi * 8 + r;
      int col = bn + wn * 64 + j * 16 + lane15;
      float v = acc[j][r] + bias[col];
      if (res) v += res[(size_t)row * N + col];
      out[(size_t)row * N + col] = v;
    }
}

// ---------------- fused FFN: x = x + ff2(relu(ff1(x))) ----------------------
__global__ __launch_bounds__(256) void k_ffn(float* __restrict__ x,
                                             const float* __restrict__ W1,
                                             const float* __restrict__ b1,
                                             const float* __restrict__ W2,
                                             const float* __restrict__ b2) {
  __shared__ half_t Xs[32 * 130];
  __shared__ half_t W1s[64 * 130];
  __shared__ half_t Hs[32 * 66];
  __shared__ half_t W2s[128 * 66];
  int tid = threadIdx.x, lane = tid & 31, w = tid >> 5;
  int bm = blockIdx.x * 32;
  int wm = w & 1, wn = w >> 1;
#pragma unroll
  for (int e = 0; e < 16; ++e) {
    int id = tid + e * 256, r = id >> 7, c = id & 127;
    Xs[r * 130 + c] = (half_t)x[(size_t)(bm + r) * EMB + c];
  }
  v8f c0 = {}, c1 = {};
  for (int ch = 0; ch < DFF / 64; ++ch) {
    int hb = ch * 64;
    __syncthreads();  // Xs ready (ch==0); W1s/W2s/Hs free of prior readers
#pragma unroll
    for (int e = 0; e < 32; ++e) {
      int id = tid + e * 256, r = id >> 7, c = id & 127;
      W1s[r * 130 + c] = (half_t)W1[(size_t)(hb + r) * EMB + c];
    }
#pragma unroll
    for (int e = 0; e < 32; ++e) {
      int id = tid + e * 256, r = id >> 6, c = id & 63;
      W2s[r * 66 + c] = (half_t)W2[(size_t)r * DFF + hb + c];
    }
    __syncthreads();
    // h(32x64) = relu(Xs @ W1chunk^T + b1)
    v8f hc = {};
    for (int k0 = 0; k0 < 128; k0 += 32) {
      v16h a, b;
      frag_a_mk(Xs, wm * 16, k0, 130, lane, a);
      frag_b_nk(W1s, wn * 16, k0, 130, lane, b);
      hc = wmma_f16(a, b, hc);
    }
    int lane15 = lane & 15, hi = lane >> 4;
#pragma unroll
    for (int r = 0; r < 8; ++r) {
      int m = wm * 16 + hi * 8 + r;
      int n = wn * 16 + lane15;
      float v = hc[r] + b1[hb + n];
      Hs[m * 66 + n] = (half_t)(v > 0.f ? v : 0.f);
    }
    __syncthreads();
    // out(32x128) += h @ W2chunk^T
    for (int k0 = 0; k0 < 64; k0 += 32) {
      v16h a, b;
      frag_a_mk(Hs, wm * 16, k0, 66, lane, a);
      frag_b_nk(W2s, wn * 32, k0, 66, lane, b);
      c0 = wmma_f16(a, b, c0);
      frag_b_nk(W2s, wn * 32 + 16, k0, 66, lane, b);
      c1 = wmma_f16(a, b, c1);
    }
  }
  int lane15 = lane & 15, hi = lane >> 4;
#pragma unroll
  for (int r = 0; r < 8; ++r) {
    int row = bm + wm * 16 + hi * 8 + r;
    int col0 = wn * 32 + lane15, col1 = col0 + 16;
    x[(size_t)row * EMB + col0] = c0[r] + b2[col0] + x[(size_t)row * EMB + col0];
    x[(size_t)row * EMB + col1] = c1[r] + b2[col1] + x[(size_t)row * EMB + col1];
  }
}

// ------------ per-molecule adjacency GEMM: x1 = relu(adj@y) + ys ------------
__global__ __launch_bounds__(256) void k_adj(const float* __restrict__ adj,
                                             const float* __restrict__ y,
                                             const float* __restrict__ ys,
                                             float* __restrict__ xout) {
  __shared__ half_t Adjs[64 * 66];
  __shared__ half_t Ys[64 * 130];
  int tid = threadIdx.x, lane = tid & 31, w = tid >> 5;
  int mol = blockIdx.x;
  const float* adjb = adj + (size_t)mol * (NAT * NAT);
  int rowbase = mol * NAT;
#pragma unroll
  for (int e = 0; e < 16; ++e) {
    int id = tid + e * 256, r = id >> 6, c = id & 63;
    Adjs[r * 66 + c] = (half_t)adjb[r * NAT + c];
  }
#pragma unroll
  for (int e = 0; e < 32; ++e) {
    int id = tid + e * 256, r = id >> 7, c = id & 127;
    Ys[r * 130 + c] = (half_t)y[(size_t)(rowbase + r) * EMB + c];
  }
  __syncthreads();
  int mt = w >> 1;         // 0..3 : 16-row tile
  int nb = (w & 1) * 64;   // column half
  v8f acc[4] = {{}, {}, {}, {}};
  for (int k0 = 0; k0 < 64; k0 += 32) {
    v16h a;
    frag_a_mk(Adjs, mt * 16, k0, 66, lane, a);
#pragma unroll
    for (int j = 0; j < 4; ++j) {
      v16h b;
      frag_b_kn(Ys, nb + j * 16, k0, 130, lane, b);
      acc[j] = wmma_f16(a, b, acc[j]);
    }
  }
  int lane15 = lane & 15, hi = lane >> 4;
#pragma unroll
  for (int j = 0; j < 4; ++j)
#pragma unroll
    for (int r = 0; r < 8; ++r) {
      int row = rowbase + mt * 16 + hi * 8 + r;
      int col = nb + j * 16 + lane15;
      float z = acc[j][r];
      z = z > 0.f ? z : 0.f;
      xout[(size_t)row * EMB + col] = z + ys[(size_t)row * EMB + col];
    }
}

// ---------------- small VALU kernels ----------------
__global__ void k_pe(const float* __restrict__ ev, const float* __restrict__ el,
                     const float* __restrict__ Wa, const float* __restrict__ ba,
                     float* __restrict__ x) {
  int tid = blockIdx.x * blockDim.x + threadIdx.x;
  if (tid >= TT * EMB) return;
  int t = tid >> 7, d = tid & 127;
  float a = ev[t]; if (a != a) a = 0.f;
  float b = el[t]; if (b != b) b = 0.f;
  x[tid] = Wa[d * 2] * a + Wa[d * 2 + 1] * b + ba[d];
}

__global__ void k_attn(const float* __restrict__ qkv,
                       const float* __restrict__ evec,
                       float* __restrict__ ao) {
  int tid = blockIdx.x * blockDim.x + threadIdx.x;
  if (tid >= BN * NHEAD * MF) return;
  int bn = tid / (NHEAD * MF);
  int rem = tid % (NHEAD * MF);
  int h = rem / MF, qm = rem % MF;
  const float* qp = qkv + (size_t)(bn * MF + qm) * 384 + h * HD;
  float s[MF];
  float smax = -3.4e38f;
#pragma unroll
  for (int km = 0; km < MF; ++km) {
    float e = evec[bn * MF + km];
    if (e != e) { s[km] = -__builtin_huge_valf(); continue; }
    const float* kp = qkv + (size_t)(bn * MF + km) * 384 + EMB + h * HD;
    float acc = 0.f;
#pragma unroll
    for (int d = 0; d < HD; ++d) acc += qp[d] * kp[d];
    s[km] = acc * 0.17677669529663687f;  // 1/sqrt(32)
    smax = fmaxf(smax, s[km]);
  }
  float p[MF], den = 0.f;
#pragma unroll
  for (int km = 0; km < MF; ++km) { p[km] = expf(s[km] - smax); den += p[km]; }
  float inv = 1.f / den;
#pragma unroll
  for (int d = 0; d < HD; ++d) {
    float acc = 0.f;
#pragma unroll
    for (int km = 0; km < MF; ++km)
      acc += p[km] * qkv[(size_t)(bn * MF + km) * 384 + 2 * EMB + h * HD + d];
    ao[(size_t)(bn * MF + qm) * EMB + h * HD + d] = acc * inv;
  }
}

__global__ __launch_bounds__(256) void k_ln(float* __restrict__ x,
                                            const float* __restrict__ g,
                                            const float* __restrict__ b,
                                            int rows) {
  int w = threadIdx.x >> 5, lane = threadIdx.x & 31;
  int row = blockIdx.x * 8 + w;
  if (row >= rows) return;
  float* xr = x + (size_t)row * EMB;
  float v[4];
  float s = 0.f;
#pragma unroll
  for (int i = 0; i < 4; ++i) { v[i] = xr[lane * 4 + i]; s += v[i]; }
#pragma unroll
  for (int off = 16; off > 0; off >>= 1) s += __shfl_xor(s, off, 32);
  float mu = s * (1.f / EMB);
  float q = 0.f;
#pragma unroll
  for (int i = 0; i < 4; ++i) { float d = v[i] - mu; q += d * d; }
#pragma unroll
  for (int off = 16; off > 0; off >>= 1) q += __shfl_xor(q, off, 32);
  float rstd = rsqrtf(q * (1.f / EMB) + LN_EPS);
#pragma unroll
  for (int i = 0; i < 4; ++i) {
    int c = lane * 4 + i;
    xr[c] = (v[i] - mu) * rstd * g[c] + b[c];
  }
}

__global__ void k_lpe_sum(const float* __restrict__ x,
                          const float* __restrict__ evec,
                          float* __restrict__ lpe) {
  int tid = blockIdx.x * blockDim.x + threadIdx.x;
  if (tid >= BN * EMB) return;
  int bn = tid >> 7, d = tid & 127;
  float s = 0.f;
#pragma unroll
  for (int m = 0; m < MF; ++m) {
    float e = evec[bn * MF + m];
    if (!(e != e)) s += x[(size_t)(bn * MF + m) * EMB + d];
  }
  lpe[tid] = s;
}

__global__ void k_raw(const int* __restrict__ atoms,
                      const float* __restrict__ emb, float* __restrict__ raw) {
  int tid = blockIdx.x * blockDim.x + threadIdx.x;
  if (tid >= BN * EMB) return;
  int bn = tid >> 7, d = tid & 127;
  raw[tid] = emb[(size_t)atoms[bn] * EMB + d];
}

__global__ void k_comb(const float* __restrict__ lpe,
                       const float* __restrict__ raw, float* __restrict__ cb) {
  int tid = blockIdx.x * blockDim.x + threadIdx.x;
  if (tid >= BN * EMB) return;
  float l = lpe[tid], r = raw[tid];
  cb[tid] = l + r - r * l;
}

static __device__ __forceinline__ float selu_f(float x) {
  const float sc = 1.0507009873554805f, al = 1.6732632423543772f;
  return x > 0.f ? sc * x : sc * al * (expf(x) - 1.f);
}

__global__ void k_accum(const float* __restrict__ x1,
                        const float* __restrict__ x2, float* __restrict__ mol) {
  int tid = blockIdx.x * blockDim.x + threadIdx.x;
  if (tid >= BN * EMB) return;
  int bn = tid >> 7, d = tid & 127;
  mol[(size_t)bn * 256 + d] += selu_f(x1[tid]);
  mol[(size_t)bn * 256 + 128 + d] += selu_f(x2[tid]);
}

__global__ void k_glo(const float* __restrict__ mol,
                      const unsigned char* __restrict__ mask,
                      float* __restrict__ glo) {
  int tid = blockIdx.x * blockDim.x + threadIdx.x;
  if (tid >= BB * 256) return;
  int b = tid >> 8, c = tid & 255;
  float s = 0.f;
  for (int n = 0; n < NAT; ++n)
    if (mask[b * NAT + n]) s += mol[(size_t)(b * NAT + n) * 256 + c];
  glo[tid] = s;
}

__global__ void k_loc(const float* __restrict__ mol,
                      const unsigned char* __restrict__ mask,
                      float* __restrict__ loc) {
  int tid = blockIdx.x * blockDim.x + threadIdx.x;
  if (tid >= BN * 256) return;
  int bn = tid >> 8;
  loc[tid] = mask[bn] ? mol[tid] : __builtin_nanf("");
}

// ---------------- host driver ----------------
static inline int cdiv(int a, int b) { return (a + b - 1) / b; }

extern "C" void kernel_launch(void* const* d_in, const int* in_sizes, int n_in,
                              void* d_out, int out_size, void* d_ws,
                              size_t ws_size, hipStream_t stream) {
  (void)in_sizes; (void)n_in; (void)out_size; (void)ws_size;
  const int* atoms = (const int*)d_in[0];
  const float* adjs = (const float*)d_in[1];
  const float* evec = (const float*)d_in[2];
  const float* eval_ = (const float*)d_in[3];
  const unsigned char* mask = (const unsigned char*)d_in[4];
  const float* emb = (const float*)d_in[5];
  const float* linA_w = (const float*)d_in[6];
  const float* linA_b = (const float*)d_in[7];
  auto P = [&](int i) { return (const float*)d_in[i]; };
  const int TF0 = 8;   // 12 tensors per transformer layer
  const int GCN0 = 32; // 8 tensors per bond (2 layers x {W.w,W.b,Ws.w,Ws.b})

  float* ws = (float*)d_ws;
  float* X    = ws;                           // [TT,128]
  float* QKV  = X + (size_t)TT * EMB;         // [TT,384]
  float* AO   = QKV + (size_t)TT * 384;       // [TT,128]
  float* LPE  = AO + (size_t)TT * EMB;        // [BN,128]
  float* RAW  = LPE + (size_t)BN * EMB;
  float* COMB = RAW + (size_t)BN * EMB;
  // GCN scratch overlays QKV region (dead after LPE phase)
  float* Y   = QKV;
  float* YS  = Y + (size_t)BN * EMB;
  float* X1  = YS + (size_t)BN * EMB;
  float* X2  = X1 + (size_t)BN * EMB;
  float* MOL = X2 + (size_t)BN * EMB;          // [BN,256]

  float* out = (float*)d_out;
  float* glo_raw = out;
  float* glo_lpe = out + (size_t)BB * 256;
  float* glo_cmb = out + (size_t)2 * BB * 256;
  float* loc     = out + (size_t)3 * BB * 256;

  // ---- raw embeddings (independent) ----
  k_raw<<<cdiv(BN * EMB, 256), 256, 0, stream>>>(atoms, emb, RAW);

  // ---- LPE transformer ----
  k_pe<<<cdiv(TT * EMB, 256), 256, 0, stream>>>(evec, eval_, linA_w, linA_b, X);
  for (int L = 0; L < 2; ++L) {
    int t = TF0 + L * 12;
    k_gemm<<<dim3(TT / 64, 3), 256, 0, stream>>>(X, P(t + 0), P(t + 1), nullptr,
                                                 QKV, TT, 384, EMB);
    k_attn<<<cdiv(BN * NHEAD * MF, 256), 256, 0, stream>>>(QKV, evec, AO);
    k_gemm<<<dim3(TT / 64, 1), 256, 0, stream>>>(AO, P(t + 2), P(t + 3), X, X,
                                                 TT, EMB, EMB);
    k_ln<<<TT / 8, 256, 0, stream>>>(X, P(t + 4), P(t + 5), TT);
    k_ffn<<<TT / 32, 256, 0, stream>>>(X, P(t + 6), P(t + 7), P(t + 8),
                                       P(t + 9));
    k_ln<<<TT / 8, 256, 0, stream>>>(X, P(t + 10), P(t + 11), TT);
  }
  k_lpe_sum<<<cdiv(BN * EMB, 256), 256, 0, stream>>>(X, evec, LPE);
  k_comb<<<cdiv(BN * EMB, 256), 256, 0, stream>>>(LPE, RAW, COMB);

  // ---- GNN over {raw, lpe, comb} ----
  const float* vins[3] = {RAW, LPE, COMB};
  float* glos[3] = {glo_raw, glo_lpe, glo_cmb};
  for (int vi = 0; vi < 3; ++vi) {
    (void)hipMemsetAsync(MOL, 0, (size_t)BN * 256 * sizeof(float), stream);
    for (int bond = 0; bond < NBOND; ++bond) {
      int g = GCN0 + bond * 8;
      const float* adj_b = adjs + (size_t)bond * BB * NAT * NAT;
      // layer 0
      k_gemm<<<dim3(BN / 64, 1), 256, 0, stream>>>(vins[vi], P(g + 0), P(g + 1),
                                                   nullptr, Y, BN, EMB, EMB);
      k_gemm<<<dim3(BN / 64, 1), 256, 0, stream>>>(vins[vi], P(g + 2), P(g + 3),
                                                   nullptr, YS, BN, EMB, EMB);
      k_adj<<<BB, 256, 0, stream>>>(adj_b, Y, YS, X1);
      // layer 1
      k_gemm<<<dim3(BN / 64, 1), 256, 0, stream>>>(X1, P(g + 4), P(g + 5),
                                                   nullptr, Y, BN, EMB, EMB);
      k_gemm<<<dim3(BN / 64, 1), 256, 0, stream>>>(X1, P(g + 6), P(g + 7),
                                                   nullptr, YS, BN, EMB, EMB);
      k_adj<<<BB, 256, 0, stream>>>(adj_b, Y, YS, X2);
      k_accum<<<cdiv(BN * EMB, 256), 256, 0, stream>>>(X1, X2, MOL);
    }
    k_glo<<<cdiv(BB * 256, 256), 256, 0, stream>>>(MOL, mask, glos[vi]);
    if (vi == 2) k_loc<<<cdiv(BN * 256, 256), 256, 0, stream>>>(MOL, mask, loc);
  }
}
